// ModelGAT_15728170238621
// MI455X (gfx1250) — matrix operand order repaired and checked by reading, not compiled, not run on hardware
//
#include <hip/hip_runtime.h>
#include <hip/hip_bf16.h>
#include <math.h>

typedef __attribute__((ext_vector_type(16))) __bf16 v16bf;
typedef __attribute__((ext_vector_type(8)))  __bf16 v8bf;
typedef __attribute__((ext_vector_type(8)))  float  v8f;

// ---------------------------------------------------------------- helpers
__device__ __forceinline__ __bf16 tobf(float f) {
    union { float f; unsigned u; } in; in.f = f;
    unsigned r = in.u + 0x7FFFu + ((in.u >> 16) & 1u);   // round-to-nearest-even
    union { unsigned short s; __bf16 b; } out;
    out.s = (unsigned short)(r >> 16);
    return out.b;
}

__device__ __forceinline__ v16bf cat16(v8bf lo, v8bf hi) {
    return __builtin_shufflevector(lo, hi, 0, 1, 2, 3, 4, 5, 6, 7,
                                           8, 9, 10, 11, 12, 13, 14, 15);
}

__device__ __forceinline__ float atomicMaxF(float* addr, float val) {
    int* ia = (int*)addr;
    int old = __float_as_int(*addr);
    while (__int_as_float(old) < val) {
        int assumed = old;
        old = atomicCAS(ia, assumed, __float_as_int(val));
        if (old == assumed) break;
    }
    return __int_as_float(old);
}

__device__ __forceinline__ float gelu_exact(float x) {
    return 0.5f * x * (1.0f + erff(x * 0.70710678118654752f));
}

// ---------------------------------------------------------------- fill
__global__ void fill_kernel(float* p, float v, long long n) {
    long long i = (long long)blockIdx.x * blockDim.x + threadIdx.x;
    if (i < n) p[i] = v;
}

// ---------------------------------------------- bf16 staging (pad + convert)
__global__ void convert_x_kernel(const float* __restrict__ X,
                                 __bf16* __restrict__ Xb, int N, int Npad) {
    int i = blockIdx.x * blockDim.x + threadIdx.x;   // Npad*64 threads
    if (i >= Npad * 64) return;
    int row = i >> 6, k = i & 63;
    float v = (row < N && k < 54) ? X[(size_t)row * 54 + k] : 0.0f;
    Xb[i] = tobf(v);
}

// W [54,128] fp32 -> Wt [128][64] bf16, K contiguous per column, zero-padded
__global__ void convert_w_kernel(const float* __restrict__ W,
                                 __bf16* __restrict__ Wt) {
    int i = blockIdx.x * blockDim.x + threadIdx.x;   // 128*64 threads
    if (i >= 128 * 64) return;
    int col = i >> 6, k = i & 63;
    float v = (k < 54) ? W[(size_t)k * 128 + col] : 0.0f;
    Wt[i] = tobf(v);
}

// ------------------------------------------------- degree + loop-attr sums
__global__ void deg_loop_kernel(const int* __restrict__ ei,
                                const float* __restrict__ eattr,
                                float* __restrict__ deg,
                                float* __restrict__ loop_attr,
                                int E) {
    int e = blockIdx.x * blockDim.x + threadIdx.x;
    if (e >= E) return;
    int dst = ei[E + e];
    atomicAdd(&deg[dst], 1.0f);
    const float* a = eattr + (size_t)e * 12;
    float* la = loop_attr + (size_t)dst * 12;
#pragma unroll
    for (int j = 0; j < 12; ++j) atomicAdd(&la[j], a[j]);
}

__global__ void loop_div_kernel(float* __restrict__ loop_attr,
                                const float* __restrict__ deg, int N) {
    int i = blockIdx.x * blockDim.x + threadIdx.x;
    if (i >= N * 12) return;
    float d = deg[i / 12];
    loop_attr[i] /= fmaxf(d, 1.0f);
}

// ------------------------------------------------- WMMA GEMM
// Xb [Npad][64] bf16, Wt [128][64] bf16 (K contiguous), O [Npad][128] f32.
// block = 256 thr = 8 waves; block owns rows [32*bid,+32); wave w owns cols [16w,+16).
// Branch-free: padded inputs, unguarded b128 loads, unguarded stores.
__global__ void gemm_wmma(const __bf16* __restrict__ Xb,
                          const __bf16* __restrict__ Wt,
                          const float* __restrict__ bias,
                          float* __restrict__ O) {
    const int lane = threadIdx.x & 31;
    const int wave = threadIdx.x >> 5;
    const int row0 = blockIdx.x * 32;
    const int ncol = wave * 16 + (lane & 15);     // B/C/D: N = lane%16
    const int hiA  = (lane >> 4) ? 8  : 0;        // A K-offset, upper half-wave
    const int hiB  = (lane >> 4) ? 8  : 0;        // B K-offset, upper half-wave
    v8f acc0 = {}, acc1 = {};
#pragma unroll
    for (int kc = 0; kc < 2; ++kc) {
        const int kb = kc * 32;
        // B fragment: halfs [kb+hiB, +8) and [kb+16+hiB, +8) of the transposed-weight column
        const __bf16* wp = Wt + (size_t)ncol * 64 + kb + hiB;
        v16bf b = cat16(*(const v8bf*)wp, *(const v8bf*)(wp + 16));
        // A fragments: halfs [kb+hiA, +8) and [kb+16+hiA, +8) of the row
        const __bf16* ap0 = Xb + (size_t)(row0 + (lane & 15)) * 64 + kb + hiA;
        v16bf a0 = cat16(*(const v8bf*)ap0, *(const v8bf*)(ap0 + 16));
        const __bf16* ap1 = ap0 + 16 * 64;
        v16bf a1 = cat16(*(const v8bf*)ap1, *(const v8bf*)(ap1 + 16));
        acc0 = __builtin_amdgcn_wmma_f32_16x16x32_bf16(
            false, a0, false, b, (short)0, acc0, false, false);
        acc1 = __builtin_amdgcn_wmma_f32_16x16x32_bf16(
            false, a1, false, b, (short)0, acc1, false, false);
    }
    const float bv = bias[ncol];
    const int rhi = (lane >> 4) ? 8 : 0;          // C/D: VGPR r -> M = r (+8 upper)
#pragma unroll
    for (int r = 0; r < 8; ++r) {
        O[(size_t)(row0 + r + rhi) * 128 + ncol]      = acc0[r] + bv;
        O[(size_t)(row0 + 16 + r + rhi) * 128 + ncol] = acc1[r] + bv;
    }
}

// ------------------------------------------------- edge logits + segment max
// one block (128 thr) per edge; wave h (32 lanes) reduces head h
__global__ void edge_score_kernel(const int* __restrict__ ei,
                                  const float* __restrict__ eattr,
                                  const float* __restrict__ loop_attr,
                                  const float* __restrict__ xl,
                                  const float* __restrict__ xr,
                                  const float* __restrict__ We,
                                  const float* __restrict__ att,
                                  float* __restrict__ ebuf,
                                  float* __restrict__ mbuf,
                                  int E, int Et) {
    int edge = blockIdx.x;
    if (edge >= Et) return;
    int c = threadIdx.x;                          // 0..127 = head*32 + ch
    int src, dst; const float* arow;
    if (edge < E) { src = ei[edge]; dst = ei[E + edge]; arow = eattr + (size_t)edge * 12; }
    else          { src = dst = edge - E;               arow = loop_attr + (size_t)src * 12; }
    float ea = 0.0f;
#pragma unroll
    for (int j = 0; j < 12; ++j) ea = fmaf(arow[j], We[j * 128 + c], ea);
    float g = xl[(size_t)src * 128 + c] + xr[(size_t)dst * 128 + c] + ea;
    g = (g >= 0.0f) ? g : 0.2f * g;               // LeakyReLU(0.2)
    float t = g * att[c];
#pragma unroll
    for (int off = 16; off > 0; off >>= 1) t += __shfl_down(t, off, 32);
    if ((threadIdx.x & 31) == 0) {
        int h = threadIdx.x >> 5;
        ebuf[(size_t)edge * 4 + h] = t;
        atomicMaxF(&mbuf[(size_t)dst * 4 + h], t);
    }
}

// ------------------------------------------------- exp + denominator
__global__ void edge_softmax_kernel(const int* __restrict__ ei,
                                    float* __restrict__ ebuf,
                                    const float* __restrict__ mbuf,
                                    float* __restrict__ denom,
                                    int E, int Et) {
    long long idx = (long long)blockIdx.x * blockDim.x + threadIdx.x;
    if (idx >= (long long)Et * 4) return;
    int edge = (int)(idx >> 2);
    int h = (int)(idx & 3);
    int dst = (edge < E) ? ei[E + edge] : edge - E;
    float ex = __expf(ebuf[idx] - mbuf[(size_t)dst * 4 + h]);
    ebuf[idx] = ex;
    atomicAdd(&denom[(size_t)dst * 4 + h], ex);
}

// ------------------------------------------------- alpha-weighted scatter
__global__ void edge_aggr_kernel(const int* __restrict__ ei,
                                 const float* __restrict__ ebuf,
                                 const float* __restrict__ denom,
                                 const float* __restrict__ xl,
                                 float* __restrict__ outp,
                                 int E, int Et) {
    long long idx = (long long)blockIdx.x * blockDim.x + threadIdx.x;
    if (idx >= (long long)Et * 128) return;
    int edge = (int)(idx >> 7);
    int c = (int)(idx & 127);
    int h = c >> 5;
    int src, dst;
    if (edge < E) { src = ei[edge]; dst = ei[E + edge]; }
    else          { src = dst = edge - E; }
    float alpha = ebuf[(size_t)edge * 4 + h] / denom[(size_t)dst * 4 + h];
    atomicAdd(&outp[(size_t)dst * 128 + c], alpha * xl[(size_t)src * 128 + c]);
}

// ------------------------------------------------- BN stats (per channel)
__global__ void bn_stats_kernel(const float* __restrict__ outp,
                                float* __restrict__ csum,
                                float* __restrict__ csumsq, int N) {
    int c = threadIdx.x;                          // blockDim = 128
    float s = 0.0f, ss = 0.0f;
    for (int i = blockIdx.x; i < N; i += gridDim.x) {
        float v = outp[(size_t)i * 128 + c];
        s += v; ss += v * v;
    }
    atomicAdd(&csum[c], s);
    atomicAdd(&csumsq[c], ss);
}

__global__ void bn_finalize_kernel(const float* __restrict__ csum,
                                   const float* __restrict__ csumsq,
                                   const float* __restrict__ gamma,
                                   const float* __restrict__ beta,
                                   float* __restrict__ scale,
                                   float* __restrict__ shift, float invN) {
    int c = threadIdx.x;                          // 128 threads, 1 block
    float mean = csum[c] * invN;
    float var  = csumsq[c] * invN - mean * mean;  // biased variance
    float rs   = rsqrtf(var + 1e-5f) * gamma[c];
    scale[c] = rs;
    shift[c] = beta[c] - mean * rs;
}

// ------------------------------------------------- BN + GELU + mean-pool scatter
__global__ void bn_gelu_pool_kernel(const float* __restrict__ outp,
                                    const float* __restrict__ scale,
                                    const float* __restrict__ shift,
                                    const int* __restrict__ batch,
                                    float* __restrict__ poolsum,
                                    float* __restrict__ poolcnt, int N) {
    long long idx = (long long)blockIdx.x * blockDim.x + threadIdx.x;
    if (idx >= (long long)N * 128) return;
    int i = (int)(idx >> 7);
    int c = (int)(idx & 127);
    float h = outp[idx] * scale[c] + shift[c];
    h = gelu_exact(h);
    int g = batch[i];
    atomicAdd(&poolsum[(size_t)g * 128 + c], h);
    if (c == 0) atomicAdd(&poolcnt[g], 1.0f);
}

__global__ void pool_div_kernel(const float* __restrict__ poolsum,
                                const float* __restrict__ poolcnt,
                                float* __restrict__ out, int total) {
    int idx = blockIdx.x * blockDim.x + threadIdx.x;
    if (idx >= total) return;
    out[idx] = poolsum[idx] / fmaxf(poolcnt[idx >> 7], 1.0f);
}

// ================================================================ launcher
extern "C" void kernel_launch(void* const* d_in, const int* in_sizes, int n_in,
                              void* d_out, int out_size, void* d_ws, size_t ws_size,
                              hipStream_t stream) {
    const float* x     = (const float*)d_in[0];
    const int*   ei    = (const int*)  d_in[1];
    const int*   batch = (const int*)  d_in[2];
    const float* eattr = (const float*)d_in[3];
    const float* Wl4   = (const float*)d_in[15];
    const float* bl4   = (const float*)d_in[16];
    const float* Wr4   = (const float*)d_in[17];
    const float* br4   = (const float*)d_in[18];
    const float* We4   = (const float*)d_in[19];
    const float* att4  = (const float*)d_in[20];
    const float* g4    = (const float*)d_in[28];
    const float* be4   = (const float*)d_in[29];

    const int N    = in_sizes[0] / 54;
    const int E    = in_sizes[1] / 2;
    const int G    = out_size / 128;
    const int Et   = E + N;
    const int Npad = (N + 31) & ~31;              // GEMM block covers 32 rows

    // scratch layout (floats first, then bf16 staging area)
    float* ws = (float*)d_ws;
    size_t off = 0;
    float* xl        = ws + off; off += (size_t)Npad * 128;
    float* xr        = ws + off; off += (size_t)Npad * 128;
    float* outp      = ws + off; off += (size_t)N * 128;
    float* ebuf      = ws + off; off += (size_t)Et * 4;
    float* mbuf      = ws + off; off += (size_t)N * 4;
    float* denom     = ws + off; off += (size_t)N * 4;   // zero-region start
    float* deg       = ws + off; off += (size_t)N;
    float* loop_attr = ws + off; off += (size_t)N * 12;
    float* csum      = ws + off; off += 128;
    float* csumsq    = ws + off; off += 128;
    float* scale     = ws + off; off += 128;
    float* shift     = ws + off; off += 128;
    float* poolsum   = ws + off; off += (size_t)G * 128;
    float* poolcnt   = ws + off; off += (size_t)G;       // zero-region end
    __bf16* Xb  = (__bf16*)(ws + off);                   // Npad*64 halfs
    __bf16* Wlt = Xb + (size_t)Npad * 64;                // 128*64 halfs
    __bf16* Wrt = Wlt + 128 * 64;
    (void)ws_size; (void)n_in;

    const int TB = 256;
    auto blocks = [](long long n, int tb) { return (unsigned)((n + tb - 1) / tb); };

    // init
    long long n_outp = (long long)N * 128;
    fill_kernel<<<blocks(n_outp, TB), TB, 0, stream>>>(outp, 0.0f, n_outp);
    long long n_m = (long long)N * 4;
    fill_kernel<<<blocks(n_m, TB), TB, 0, stream>>>(mbuf, -1e30f, n_m);
    long long n_zero = (long long)(poolcnt + G - denom); // denom..poolcnt contiguous
    fill_kernel<<<blocks(n_zero, TB), TB, 0, stream>>>(denom, 0.0f, n_zero);

    // bf16 staging: padded X and K-major transposed weights
    convert_x_kernel<<<blocks((long long)Npad * 64, TB), TB, 0, stream>>>(x, Xb, N, Npad);
    convert_w_kernel<<<blocks(128 * 64, TB), TB, 0, stream>>>(Wl4, Wlt);
    convert_w_kernel<<<blocks(128 * 64, TB), TB, 0, stream>>>(Wr4, Wrt);

    // self-loop edge_attr fill (mean of incoming)
    deg_loop_kernel<<<blocks(E, TB), TB, 0, stream>>>(ei, eattr, deg, loop_attr, E);
    loop_div_kernel<<<blocks((long long)N * 12, TB), TB, 0, stream>>>(loop_attr, deg, N);

    // node transforms via WMMA (bf16 in, f32 accum), branch-free inner loop
    unsigned gemm_blocks = (unsigned)(Npad / 32);
    gemm_wmma<<<gemm_blocks, 256, 0, stream>>>(Xb, Wlt, bl4, xl);
    gemm_wmma<<<gemm_blocks, 256, 0, stream>>>(Xb, Wrt, br4, xr);

    // attention logits + segment max
    edge_score_kernel<<<(unsigned)Et, 128, 0, stream>>>(ei, eattr, loop_attr, xl, xr,
                                                        We4, att4, ebuf, mbuf, E, Et);
    // softmax numerator + denominator
    edge_softmax_kernel<<<blocks((long long)Et * 4, TB), TB, 0, stream>>>(
        ei, ebuf, mbuf, denom, E, Et);
    // weighted scatter-aggregate
    edge_aggr_kernel<<<blocks((long long)Et * 128, TB), TB, 0, stream>>>(
        ei, ebuf, denom, xl, outp, E, Et);

    // batch-norm (b4 cancels under mean subtraction, omitted) + GELU + pool
    bn_stats_kernel<<<256, 128, 0, stream>>>(outp, csum, csumsq, N);
    bn_finalize_kernel<<<1, 128, 0, stream>>>(csum, csumsq, g4, be4, scale, shift,
                                              1.0f / (float)N);
    bn_gelu_pool_kernel<<<blocks((long long)N * 128, TB), TB, 0, stream>>>(
        outp, scale, shift, batch, poolsum, poolcnt, N);
    pool_div_kernel<<<blocks(out_size, TB), TB, 0, stream>>>(
        poolsum, poolcnt, (float*)d_out, out_size);
}
